// LLama3_1_20272245637482
// MI455X (gfx1250) — matrix operand order repaired and checked by reading, not compiled, not run on hardware
//
#include <hip/hip_runtime.h>

#define S_   2048
#define D_   2048
#define H_   16
#define KVH_ 4
#define HD_  128
#define F_   8192
#define EPS_ 1e-5f

typedef __attribute__((ext_vector_type(16))) __bf16 v16bf;
typedef __attribute__((ext_vector_type(8)))  float  v8f;

union FragAB { v16bf v; uint4 u[2]; };

__device__ __forceinline__ unsigned short f2bf(float f) {
  unsigned int u = __float_as_uint(f);
  u += 0x7fffu + ((u >> 16) & 1u);
  return (unsigned short)(u >> 16);
}

__device__ __forceinline__ v8f v8zero() {
  v8f z;
#pragma unroll
  for (int i = 0; i < 8; ++i) z[i] = 0.0f;
  return z;
}

// CDNA5 async copy: global -> LDS, 16 bytes per lane, tracked by ASYNCcnt.
__device__ __forceinline__ void async_copy_b128(unsigned lds_off, const void* gptr) {
  asm volatile("global_load_async_to_lds_b128 %0, %1, off"
               :: "v"(lds_off), "v"(gptr) : "memory");
}
__device__ __forceinline__ void wait_async0() {
  asm volatile("s_wait_asynccnt 0" ::: "memory");
}
__device__ __forceinline__ unsigned lds_off_of(const void* p) {
  // generic LDS address carries the LDS byte offset in its low 32 bits
  return (unsigned)(unsigned long long)p;
}

// ---------------------------------------------------------------------------
// GEMM: C[M,N] (f32) = A[M,K] (bf16, row-major) * B[N,K]^T (f32 row-major,
// converted to bf16 during LDS staging) + optional residual.
// Block tile 128x128, K-step 32, 8 waves (2x4), wave tile 64x32.
// Double-buffered LDS; A tiles copied with global_load_async_to_lds_b128,
// B tiles loaded to regs before compute and converted/stored after compute.
// ---------------------------------------------------------------------------
#define BM 128
#define BN 128
#define BK 32
#define LDT (BK + 8)

__global__ __launch_bounds__(256) void gemm_bf16f32(
    const unsigned short* __restrict__ A,
    const float* __restrict__ B,
    const float* __restrict__ res,
    float* __restrict__ C,
    int M, int N, int K)
{
  __shared__ __align__(16) unsigned short As[2][BM][LDT];
  __shared__ __align__(16) unsigned short Bs[2][BN][LDT];

  const int tid  = threadIdx.x;
  const int lane = tid & 31;
  const int wave = tid >> 5;
  const int wm   = wave & 1;   // 0..1  (64-row slab)
  const int wn   = wave >> 1;  // 0..3  (32-col slab)
  const int m0   = blockIdx.y * BM;
  const int n0   = blockIdx.x * BN;

  v8f acc[4][2];
#pragma unroll
  for (int i = 0; i < 4; ++i)
#pragma unroll
    for (int j = 0; j < 2; ++j) acc[i][j] = v8zero();

  const int lr = tid >> 2;         // 0..63
  const int lc = (tid & 3) * 8;    // 0,8,16,24

  float4 breg[2][2];

  auto issueA = [&](int buf, int k0) {
#pragma unroll
    for (int hh = 0; hh < 2; ++hh) {
      int r = lr + hh * 64;
      async_copy_b128(lds_off_of(&As[buf][r][lc]),
                      A + (size_t)(m0 + r) * K + k0 + lc);
    }
  };
  auto loadB = [&](int k0) {
#pragma unroll
    for (int hh = 0; hh < 2; ++hh) {
      int r = lr + hh * 64;
      const float* bp = B + (size_t)(n0 + r) * K + k0 + lc;
      breg[hh][0] = *(const float4*)bp;
      breg[hh][1] = *(const float4*)(bp + 4);
    }
  };
  auto storeB = [&](int buf) {
#pragma unroll
    for (int hh = 0; hh < 2; ++hh) {
      int r = lr + hh * 64;
      union { unsigned short s[8]; uint4 u; } t;
      t.s[0] = f2bf(breg[hh][0].x); t.s[1] = f2bf(breg[hh][0].y);
      t.s[2] = f2bf(breg[hh][0].z); t.s[3] = f2bf(breg[hh][0].w);
      t.s[4] = f2bf(breg[hh][1].x); t.s[5] = f2bf(breg[hh][1].y);
      t.s[6] = f2bf(breg[hh][1].z); t.s[7] = f2bf(breg[hh][1].w);
      *(uint4*)&Bs[buf][r][lc] = t.u;
    }
  };

  const int nk = K / BK;
  issueA(0, 0);
  loadB(0);
  storeB(0);

  int buf = 0;
  for (int kt = 0; kt < nk; ++kt) {
    wait_async0();
    __syncthreads();

    const int nbuf = buf ^ 1;
    const bool hasNext = (kt + 1 < nk);
    if (hasNext) {
      issueA(nbuf, (kt + 1) * BK);
      loadB((kt + 1) * BK);
    }

    const int khalfA = (lane & 16) ? 8 : 0;    // A: lanes16-31 hold K=8..15 / 24..31
    const int kbaseB = (lane & 16) ? 8 : 0;    // B: lanes16-31 hold K=8..15 / 24..31
    FragAB bF[2];
#pragma unroll
    for (int nt = 0; nt < 2; ++nt) {
      int n = wn * 32 + nt * 16 + (lane & 15);
      bF[nt].u[0] = *(const uint4*)&Bs[buf][n][kbaseB];
      bF[nt].u[1] = *(const uint4*)&Bs[buf][n][kbaseB + 16];
    }
#pragma unroll
    for (int mt = 0; mt < 4; ++mt) {
      int r = wm * 64 + mt * 16 + (lane & 15);
      FragAB aF;
      aF.u[0] = *(const uint4*)&As[buf][r][khalfA];
      aF.u[1] = *(const uint4*)&As[buf][r][16 + khalfA];
#pragma unroll
      for (int nt = 0; nt < 2; ++nt) {
        acc[mt][nt] = __builtin_amdgcn_wmma_f32_16x16x32_bf16(
            false, aF.v, false, bF[nt].v, (short)0, acc[mt][nt], false, false);
      }
    }

    if (hasNext) storeB(nbuf);
    buf = nbuf;
  }

  if (res != nullptr) {
#pragma unroll
    for (int mt = 0; mt < 4; ++mt) {
      int rbase = m0 + wm * 64 + mt * 16 + ((lane & 16) ? 8 : 0);
#pragma unroll
      for (int nt = 0; nt < 2; ++nt) {
        int n = n0 + wn * 32 + nt * 16 + (lane & 15);
#pragma unroll
        for (int r = 0; r < 8; ++r) {
          size_t idx = (size_t)(rbase + r) * N + n;
          C[idx] = acc[mt][nt][r] + res[idx];
        }
      }
    }
  } else {
#pragma unroll
    for (int mt = 0; mt < 4; ++mt) {
      int rbase = m0 + wm * 64 + mt * 16 + ((lane & 16) ? 8 : 0);
#pragma unroll
      for (int nt = 0; nt < 2; ++nt) {
        int n = n0 + wn * 32 + nt * 16 + (lane & 15);
#pragma unroll
        for (int r = 0; r < 8; ++r) {
          size_t idx = (size_t)(rbase + r) * N + n;
          C[idx] = acc[mt][nt][r];
        }
      }
    }
  }
}

// ---------------------------------------------------------------------------
// Flash-style causal GQA attention. Grid: (S/128, H). 8 waves, each wave owns
// 16 query rows -> row softmax stats stay wave-local (C-layout row striping).
// K tile staged with async global->LDS copies.
// ---------------------------------------------------------------------------
__global__ __launch_bounds__(256) void attn_kernel(
    const unsigned short* __restrict__ Q,   // S x (H*HD) bf16
    const unsigned short* __restrict__ Km,  // S x (KVH*HD) bf16 (rope applied)
    const unsigned short* __restrict__ Vm,  // S x (KVH*HD) bf16
    unsigned short* __restrict__ O)         // S x (H*HD) bf16
{
  __shared__ __align__(16) unsigned short Kt[64][HD_ + 8];
  __shared__ __align__(16) unsigned short Vt[HD_][64 + 8];
  __shared__ __align__(16) unsigned short Pt[8][16][64 + 8];

  const int tid   = threadIdx.x;
  const int lane  = tid & 31;
  const int wave  = tid >> 5;
  const int qbase = blockIdx.x * 128;
  const int h     = blockIdx.y;
  const int kvh   = h >> 2;
  const int qr0   = qbase + wave * 16;
  const float scale = 0.08838834764831845f; // 1/sqrt(128)

  // Q fragments (A layout), 16 rows x 128 dims = 4 K-chunks of 32
  FragAB qF[4];
  {
    int row   = qr0 + (lane & 15);
    int khalf = (lane & 16) ? 8 : 0;
    const unsigned short* qp = Q + (size_t)row * (H_ * HD_) + h * HD_;
#pragma unroll
    for (int kc = 0; kc < 4; ++kc) {
      qF[kc].u[0] = *(const uint4*)(qp + kc * 32 + khalf);
      qF[kc].u[1] = *(const uint4*)(qp + kc * 32 + 16 + khalf);
    }
  }

  v8f o[8];
  float m_run[8], l_run[8];
#pragma unroll
  for (int i = 0; i < 8; ++i) { o[i] = v8zero(); m_run[i] = -3.0e38f; l_run[i] = 0.0f; }

  const int nkb = qbase / 64 + 2;   // key blocks of 64 covering causal span
  for (int kb = 0; kb < nkb; ++kb) {
    __syncthreads();
    for (int i = tid; i < 64 * 16; i += 256) {        // K tile row-major (async)
      int r = i >> 4, c = (i & 15) * 8;
      async_copy_b128(lds_off_of(&Kt[r][c]),
                      Km + (size_t)(kb * 64 + r) * (KVH_ * HD_) + kvh * HD_ + c);
    }
    for (int i = tid; i < 64 * 128; i += 256) {       // V tile transposed
      int r = i >> 7, c = i & 127;
      Vt[c][r] = Vm[(size_t)(kb * 64 + r) * (KVH_ * HD_) + kvh * HD_ + c];
    }
    wait_async0();
    __syncthreads();

    // scores: 16 x 64 in four 16x16 accumulators
    v8f sc[4];
#pragma unroll
    for (int nt = 0; nt < 4; ++nt) sc[nt] = v8zero();
    const int kselK = (lane & 16) ? 8 : 0;
#pragma unroll
    for (int nt = 0; nt < 4; ++nt) {
      int n = nt * 16 + (lane & 15);
#pragma unroll
      for (int kc = 0; kc < 4; ++kc) {
        FragAB bF;
        bF.u[0] = *(const uint4*)&Kt[n][kc * 32 + kselK];
        bF.u[1] = *(const uint4*)&Kt[n][kc * 32 + kselK + 16];
        sc[nt] = __builtin_amdgcn_wmma_f32_16x16x32_bf16(
            false, qF[kc].v, false, bF.v, (short)0, sc[nt], false, false);
      }
    }

    // online softmax (row r of this wave = qr0 + r + (lane>=16 ? 8 : 0))
    const int rofs = (lane & 16) ? 8 : 0;
    const int kcol = lane & 15;
#pragma unroll
    for (int r = 0; r < 8; ++r) {
      int qpos = qr0 + r + rofs;
      float mx = -3.0e38f;
#pragma unroll
      for (int nt = 0; nt < 4; ++nt) {
        int kpos = kb * 64 + nt * 16 + kcol;
        float sv = sc[nt][r] * scale;
        if (kpos > qpos) sv = -3.0e38f;
        sc[nt][r] = sv;
        mx = fmaxf(mx, sv);
      }
#pragma unroll
      for (int off = 1; off < 16; off <<= 1) mx = fmaxf(mx, __shfl_xor(mx, off));
      float mnew = fmaxf(m_run[r], mx);
      float fac  = __expf(m_run[r] - mnew);
      m_run[r]   = mnew;
      float rs = 0.0f;
#pragma unroll
      for (int nt = 0; nt < 4; ++nt) {
        float pv = __expf(sc[nt][r] - mnew);
        rs += pv;
        Pt[wave][r + rofs][nt * 16 + kcol] = f2bf(pv);
      }
#pragma unroll
      for (int off = 1; off < 16; off <<= 1) rs += __shfl_xor(rs, off);
      l_run[r] = l_run[r] * fac + rs;
#pragma unroll
      for (int dt = 0; dt < 8; ++dt) o[dt][r] *= fac;
    }

    // O += P @ V   (P: 16x64 bf16 via LDS round-trip to A layout)
    const int khalf = (lane & 16) ? 8 : 0;
    const int kselV = (lane & 16) ? 8 : 0;
#pragma unroll
    for (int kc = 0; kc < 2; ++kc) {
      FragAB pF;
      int pr = lane & 15;
      pF.u[0] = *(const uint4*)&Pt[wave][pr][kc * 32 + khalf];
      pF.u[1] = *(const uint4*)&Pt[wave][pr][kc * 32 + 16 + khalf];
#pragma unroll
      for (int dt = 0; dt < 8; ++dt) {
        FragAB bF;
        int d = dt * 16 + (lane & 15);
        bF.u[0] = *(const uint4*)&Vt[d][kc * 32 + kselV];
        bF.u[1] = *(const uint4*)&Vt[d][kc * 32 + kselV + 16];
        o[dt] = __builtin_amdgcn_wmma_f32_16x16x32_bf16(
            false, pF.v, false, bF.v, (short)0, o[dt], false, false);
      }
    }
  }

  const int rofs = (lane & 16) ? 8 : 0;
#pragma unroll
  for (int r = 0; r < 8; ++r) {
    float inv = 1.0f / l_run[r];
    int row = qr0 + r + rofs;
#pragma unroll
    for (int dt = 0; dt < 8; ++dt) {
      O[(size_t)row * (H_ * HD_) + h * HD_ + dt * 16 + (lane & 15)] =
          f2bf(o[dt][r] * inv);
    }
  }
}

// ---------------------------------------------------------------------------
// Elementwise kernels
// ---------------------------------------------------------------------------
__global__ __launch_bounds__(256) void rmsnorm_kernel(
    const float* __restrict__ x, const float* __restrict__ w,
    unsigned short* __restrict__ out, int D)
{
  __shared__ float red[256];
  const int row = blockIdx.x;
  const int tid = threadIdx.x;
  float ss = 0.0f;
  for (int i = tid; i < D; i += 256) { float v = x[(size_t)row * D + i]; ss += v * v; }
  red[tid] = ss;
  __syncthreads();
  for (int s = 128; s > 0; s >>= 1) {
    if (tid < s) red[tid] += red[tid + s];
    __syncthreads();
  }
  const float sc = rsqrtf(red[0] / (float)D + EPS_);
  for (int i = tid; i < D; i += 256)
    out[(size_t)row * D + i] = f2bf(x[(size_t)row * D + i] * sc * w[i]);
}

__global__ void rope_kernel(const float* __restrict__ t,
                            const float* __restrict__ fc,
                            const float* __restrict__ fs,
                            unsigned short* __restrict__ out, int nh)
{
  int idx = blockIdx.x * blockDim.x + threadIdx.x;
  int total = S_ * nh * (HD_ / 2);
  if (idx >= total) return;
  int j   = idx % (HD_ / 2);
  int rem = idx / (HD_ / 2);
  int hh  = rem % nh;
  int s   = rem / nh;
  size_t base = ((size_t)s * nh + hh) * HD_ + 2 * j;
  float t0 = t[base], t1 = t[base + 1];
  float c = fc[s * (HD_ / 2) + j], sn = fs[s * (HD_ / 2) + j];
  out[base]     = f2bf(t0 * c - t1 * sn);
  out[base + 1] = f2bf(t0 * sn + t1 * c);
}

__global__ void cvt_kernel(const float* __restrict__ in,
                           unsigned short* __restrict__ out, size_t n)
{
  size_t i = (size_t)blockIdx.x * blockDim.x + threadIdx.x;
  if (i < n) out[i] = f2bf(in[i]);
}

__global__ void silumul_kernel(const float* __restrict__ a1,
                               const float* __restrict__ a3,
                               unsigned short* __restrict__ g, size_t n)
{
  size_t i = (size_t)blockIdx.x * blockDim.x + threadIdx.x;
  if (i < n) {
    float a = a1[i];
    float s = a / (1.0f + __expf(-a));
    g[i] = f2bf(s * a3[i]);
  }
}

// ---------------------------------------------------------------------------
extern "C" void kernel_launch(void* const* d_in, const int* in_sizes, int n_in,
                              void* d_out, int out_size, void* d_ws, size_t ws_size,
                              hipStream_t stream)
{
  (void)in_sizes; (void)n_in; (void)out_size; (void)ws_size;
  const float* x0   = (const float*)d_in[0];
  const float* fcos = (const float*)d_in[1];
  const float* fsin = (const float*)d_in[2];
  const float* anw  = (const float*)d_in[3];
  const float* wq   = (const float*)d_in[4];
  const float* wk   = (const float*)d_in[5];
  const float* wv   = (const float*)d_in[6];
  const float* wo   = (const float*)d_in[7];
  const float* fnw  = (const float*)d_in[8];
  const float* w1   = (const float*)d_in[9];
  const float* w2   = (const float*)d_in[10];
  const float* w3   = (const float*)d_in[11];

  char* p = (char*)d_ws;
  auto alloc = [&](size_t bytes) -> char* {
    char* r = p;
    p += (bytes + 255) & ~(size_t)255;
    return r;
  };
  float*          xf  = (float*)         alloc((size_t)S_ * D_ * 4);
  unsigned short* h16 = (unsigned short*)alloc((size_t)S_ * D_ * 2);
  float*          q32 = (float*)         alloc((size_t)S_ * D_ * 4);
  float*          k32 = (float*)         alloc((size_t)S_ * KVH_ * HD_ * 4);
  float*          v32 = (float*)         alloc((size_t)S_ * KVH_ * HD_ * 4);
  unsigned short* q16 = (unsigned short*)alloc((size_t)S_ * D_ * 2);
  unsigned short* k16 = (unsigned short*)alloc((size_t)S_ * KVH_ * HD_ * 2);
  unsigned short* v16 = (unsigned short*)alloc((size_t)S_ * KVH_ * HD_ * 2);
  unsigned short* at16= (unsigned short*)alloc((size_t)S_ * D_ * 2);
  float*          a1  = (float*)         alloc((size_t)S_ * F_ * 4);
  float*          a3  = (float*)         alloc((size_t)S_ * F_ * 4);
  unsigned short* g16 = (unsigned short*)alloc((size_t)S_ * F_ * 2);

  hipMemcpyAsync(xf, x0, (size_t)S_ * D_ * 4, hipMemcpyDeviceToDevice, stream);

  for (int l = 0; l < 2; ++l) {
    rmsnorm_kernel<<<S_, 256, 0, stream>>>(xf, anw + (size_t)l * D_, h16, D_);

    gemm_bf16f32<<<dim3(D_ / 128, S_ / 128), 256, 0, stream>>>(
        h16, wq + (size_t)l * D_ * D_, nullptr, q32, S_, D_, D_);
    gemm_bf16f32<<<dim3((KVH_ * HD_) / 128, S_ / 128), 256, 0, stream>>>(
        h16, wk + (size_t)l * KVH_ * HD_ * D_, nullptr, k32, S_, KVH_ * HD_, D_);
    gemm_bf16f32<<<dim3((KVH_ * HD_) / 128, S_ / 128), 256, 0, stream>>>(
        h16, wv + (size_t)l * KVH_ * HD_ * D_, nullptr, v32, S_, KVH_ * HD_, D_);

    int nq = S_ * H_ * (HD_ / 2);
    rope_kernel<<<(nq + 255) / 256, 256, 0, stream>>>(q32, fcos, fsin, q16, H_);
    int nk = S_ * KVH_ * (HD_ / 2);
    rope_kernel<<<(nk + 255) / 256, 256, 0, stream>>>(k32, fcos, fsin, k16, KVH_);
    size_t nv = (size_t)S_ * KVH_ * HD_;
    cvt_kernel<<<(unsigned)((nv + 255) / 256), 256, 0, stream>>>(v32, v16, nv);

    attn_kernel<<<dim3(S_ / 128, H_), 256, 0, stream>>>(q16, k16, v16, at16);

    gemm_bf16f32<<<dim3(D_ / 128, S_ / 128), 256, 0, stream>>>(
        at16, wo + (size_t)l * D_ * D_, xf, xf, S_, D_, D_);

    rmsnorm_kernel<<<S_, 256, 0, stream>>>(xf, fnw + (size_t)l * D_, h16, D_);

    gemm_bf16f32<<<dim3(F_ / 128, S_ / 128), 256, 0, stream>>>(
        h16, w1 + (size_t)l * F_ * D_, nullptr, a1, S_, F_, D_);
    gemm_bf16f32<<<dim3(F_ / 128, S_ / 128), 256, 0, stream>>>(
        h16, w3 + (size_t)l * F_ * D_, nullptr, a3, S_, F_, D_);

    size_t ng = (size_t)S_ * F_;
    silumul_kernel<<<(unsigned)((ng + 255) / 256), 256, 0, stream>>>(a1, a3, g16, ng);

    gemm_bf16f32<<<dim3(D_ / 128, S_ / 128), 256, 0, stream>>>(
        g16, w2 + (size_t)l * D_ * F_, xf, xf, S_, D_, F_);
  }

  hipMemcpyAsync(d_out, xf, (size_t)S_ * D_ * 4, hipMemcpyDeviceToDevice, stream);
}
